// LeWinAttention_46866683134119
// MI455X (gfx1250) — matrix operand order repaired and checked by reading, not compiled, not run on hardware
//
#include <hip/hip_runtime.h>
#include <hip/hip_bf16.h>

// LeWin / Swin window attention for gfx1250 (MI455X), wave32 + WMMA f16.
// Pre-pass packs fp32 weights into f16 fragment-major layout in d_ws (once
// per launch, amortized over 8192 blocks) and materializes the rel-pos bias.
// Main kernel: one workgroup (8 waves) per window, all GEMMs on
// v_wmma_f32_16x16x32_f16, fp32 softmax, register-resident proj accumulators.

#define DIM       256
#define NHEADS    8
#define HEADDIM   32
#define NTOK      64
#define XS_STRIDE 264   // f16 elems per row of x tile (256 + pad)
#define QSTR      40    // f16 elems per row of q/k/v/oh tiles (32 + pad)
#define SSTR      68    // f32 elems per row of score tile (64 + pad)

#define SMEM_XS   0
#define SMEM_QH   33792                  // 64*264*2
#define SMEM_KH   (SMEM_QH + 5120)      // 64*40*2
#define SMEM_VH   (SMEM_KH + 5120)
#define SMEM_S    (SMEM_VH + 5120)
#define SMEM_OH   (SMEM_S + 17408)      // 64*68*4
#define SMEM_TOT  (SMEM_OH + 5120)      // = 71680 bytes

// d_ws layout (bytes)
#define QKV_FRAGS   384                  // mat(3) x head(8) x ntile(2) x kstep(8)
#define PROJ_FRAGS  128                  // ntile(16) x head(8)
#define FRAG_BYTES  1024                 // 32 lanes x 16 f16 x 2B
#define WS_QKV_OFF  0
#define WS_PROJ_OFF (QKV_FRAGS * FRAG_BYTES)             // 393216
#define WS_BIAS_OFF (WS_PROJ_OFF + PROJ_FRAGS * FRAG_BYTES) // 524288
#define WS_TOTAL    (WS_BIAS_OFF + NHEADS * NTOK * NTOK * 4) // 655360

typedef _Float16 h16;
typedef __attribute__((ext_vector_type(16))) _Float16 v16h;
typedef __attribute__((ext_vector_type(8)))  float    v8f;

// ---- fragment loaders, contiguous k order (k = 16*half + s), same for A and B ------------------------
// A (16xK=32, f16): lane = M (0..15), lanes 0-15 hold K=0..15, 16-31 hold 16..31.
__device__ __forceinline__ int ka_off(int i, int half) {
    return half * 16 + 2 * i;
}
// B (K=32x16, f16): lane = N (0..15), lanes 0-15 hold K=0..15, 16-31 hold 16..31.
__device__ __forceinline__ int kb_off(int i, int half) {
    return half * 16 + 2 * i;
}

// A operand from f16 [M][K] storage
__device__ __forceinline__ v16h load_a_h(const h16* p, int stride, int lane) {
    int r = lane & 15, hf = lane >> 4;
    v16h a;
#pragma unroll
    for (int i = 0; i < 8; ++i) {
        int k = ka_off(i, hf);
        a[2 * i]     = p[r * stride + k];
        a[2 * i + 1] = p[r * stride + k + 1];
    }
    return a;
}
// A operand from f32 [M][K] storage (convert)
__device__ __forceinline__ v16h load_a_f(const float* p, int stride, int lane) {
    int r = lane & 15, hf = lane >> 4;
    v16h a;
#pragma unroll
    for (int i = 0; i < 8; ++i) {
        int k = ka_off(i, hf);
        a[2 * i]     = (h16)p[r * stride + k];
        a[2 * i + 1] = (h16)p[r * stride + k + 1];
    }
    return a;
}
// B operand from f16 [N][K] storage (k-matrix in LDS)
__device__ __forceinline__ v16h load_b_nk_h(const h16* p, int stride, int lane) {
    int n = lane & 15, hf = lane >> 4;
    v16h b;
#pragma unroll
    for (int i = 0; i < 8; ++i) {
        int k = kb_off(i, hf);
        b[2 * i]     = p[n * stride + k];
        b[2 * i + 1] = p[n * stride + k + 1];
    }
    return b;
}
// B operand from f16 [K][N] storage (v-matrix in LDS)
__device__ __forceinline__ v16h load_b_kn_h(const h16* p, int stride, int lane) {
    int n = lane & 15, hf = lane >> 4;
    v16h b;
#pragma unroll
    for (int i = 0; i < 8; ++i) {
        int k = kb_off(i, hf);
        b[2 * i]     = p[k * stride + n];
        b[2 * i + 1] = p[(k + 1) * stride + n];
    }
    return b;
}

#define WMMA(A, B, C) \
    __builtin_amdgcn_wmma_f32_16x16x32_f16(false, (A), false, (B), (short)0, (C), false, false)

// ---------------------------------------------------------------------------
// Pre-pass: pack qkv_w / proj_w into f16 fragment-major layout + bias matrix.
// qkv fragment id  = mat*128 + h*16 + nt*8 + ks  (1 KB each; lane n=l&15,
//   hf=l>>4 holds 16 contiguous f16 = W[row = mat*256+h*32+nt*16+n]
//                                      [col = ks*32 + hf*16 .. +16))
// proj fragment id = nt*8 + h  (lane holds W[nt*16+n][h*32 + hf*16 .. +16))
// bias[h*4096 + n*64 + m] = rel_bias_table[rel_pos_index[n][m]][h]
// ---------------------------------------------------------------------------
__global__ void __launch_bounds__(256)
pack_kernel(const float* __restrict__ qkv_w,
            const float* __restrict__ proj_w,
            const float* __restrict__ bias_table,
            const int*   __restrict__ rel_idx,
            void* __restrict__ ws) {
    int tid = blockIdx.x * 256 + threadIdx.x;
    h16*   qpack = (h16*)((char*)ws + WS_QKV_OFF);
    h16*   ppack = (h16*)((char*)ws + WS_PROJ_OFF);
    float* biasp = (float*)((char*)ws + WS_BIAS_OFF);

    if (tid < QKV_FRAGS * 32) {
        int fid = tid >> 5, lane = tid & 31;
        int ks = fid & 7, nt = (fid >> 3) & 1, hh = (fid >> 4) & 7, mat = fid >> 7;
        int row  = mat * DIM + hh * HEADDIM + nt * 16 + (lane & 15);
        int col0 = ks * 32 + (lane >> 4) * 16;
        const float4* src = (const float4*)(qkv_w + (size_t)row * DIM + col0);
        v16h frag;
#pragma unroll
        for (int j = 0; j < 4; ++j) {
            float4 v4 = src[j];
            frag[4 * j]     = (h16)v4.x;
            frag[4 * j + 1] = (h16)v4.y;
            frag[4 * j + 2] = (h16)v4.z;
            frag[4 * j + 3] = (h16)v4.w;
        }
        ((v16h*)qpack)[fid * 32 + lane] = frag;
    } else if (tid < (QKV_FRAGS + PROJ_FRAGS) * 32) {
        int t2 = tid - QKV_FRAGS * 32;
        int fid = t2 >> 5, lane = t2 & 31;
        int hh = fid & 7, nt = fid >> 3;
        int row  = nt * 16 + (lane & 15);
        int col0 = hh * HEADDIM + (lane >> 4) * 16;
        const float4* src = (const float4*)(proj_w + (size_t)row * DIM + col0);
        v16h frag;
#pragma unroll
        for (int j = 0; j < 4; ++j) {
            float4 v4 = src[j];
            frag[4 * j]     = (h16)v4.x;
            frag[4 * j + 1] = (h16)v4.y;
            frag[4 * j + 2] = (h16)v4.z;
            frag[4 * j + 3] = (h16)v4.w;
        }
        ((v16h*)ppack)[fid * 32 + lane] = frag;
    } else {
        int e = tid - (QKV_FRAGS + PROJ_FRAGS) * 32;   // 0 .. 32767
        if (e < NHEADS * NTOK * NTOK) {
            int hh = e >> 12, rc = e & 4095;
            biasp[e] = bias_table[rel_idx[rc] * NHEADS + hh];
        }
    }
}

// ---------------------------------------------------------------------------
// Main attention kernel
// ---------------------------------------------------------------------------
__global__ void __launch_bounds__(256)
lewin_attn_kernel(const float* __restrict__ x,
                  const float* __restrict__ qkv_b,
                  const float* __restrict__ proj_b,
                  const void*  __restrict__ ws,
                  float* __restrict__ out) {
    extern __shared__ char smem[];
    h16*   xs = (h16*)(smem + SMEM_XS);   // [64][264] x in f16
    h16*   qh = (h16*)(smem + SMEM_QH);   // [64][40]  q for current head
    h16*   kh = (h16*)(smem + SMEM_KH);   // [64][40]  k
    h16*   vh = (h16*)(smem + SMEM_VH);   // [64][40]  v
    float* S  = (float*)(smem + SMEM_S);  // [64][68]  scores / probs (f32)
    h16*   oh = (h16*)(smem + SMEM_OH);   // [64][40]  head output

    const v16h*  qpack = (const v16h*)((const char*)ws + WS_QKV_OFF);
    const v16h*  ppack = (const v16h*)((const char*)ws + WS_PROJ_OFF);
    const float* biasp = (const float*)((const char*)ws + WS_BIAS_OFF);

    const int b    = blockIdx.x;
    const int t    = threadIdx.x;
    const int lane = t & 31;
    const int w    = t >> 5;              // wave id 0..7

    // Register-resident projection accumulators: wave w owns M-tile (w&3),
    // N-tiles [(w>>2)*8 .. +8) of the 64x256 output.
    v8f pacc[8];
#pragma unroll
    for (int j = 0; j < 8; ++j) pacc[j] = {};

    // ---- stage 0: x tile -> LDS f16 -------------------------------------
    const float* xb = x + (size_t)b * NTOK * DIM;
#pragma unroll
    for (int i = 0; i < 16; ++i) {
        int e = t + i * 256;              // float4 index, 4096 total
        float4 v4 = ((const float4*)xb)[e];
        int row = e >> 6;                 // 64 float4 per row
        int c4  = (e & 63) * 4;
        h16* d = xs + row * XS_STRIDE + c4;
        d[0] = (h16)v4.x; d[1] = (h16)v4.y; d[2] = (h16)v4.z; d[3] = (h16)v4.w;
    }
    __syncthreads();

    const float scale = 0.17677669529663687f; // 1/sqrt(32)

#pragma unroll 1
    for (int h = 0; h < NHEADS; ++h) {
        // ---- stage 1: q/k/v for this head (24 tiles, 3 per wave) --------
#pragma unroll
        for (int s = 0; s < 3; ++s) {
            int tt  = w * 3 + s;          // 0..23
            int mat = tt >> 3;            // 0=q 1=k 2=v
            int sub = tt & 7;
            int mt  = sub >> 1, nt = sub & 1;
            int fidbase = mat * 128 + h * 16 + nt * 8;
            v8f acc = {};
#pragma unroll
            for (int ks = 0; ks < 8; ++ks) {
                v16h A = load_a_h(xs + mt * 16 * XS_STRIDE + ks * 32, XS_STRIDE, lane);
                v16h B = qpack[(fidbase + ks) * 32 + lane];   // packed f16, coalesced
                acc = WMMA(A, B, acc);
            }
            h16* dst = (mat == 0) ? qh : (mat == 1) ? kh : vh;
            int hf = lane >> 4, colg = nt * 16 + (lane & 15);
            float bv = qkv_b[mat * DIM + h * HEADDIM + colg];
#pragma unroll
            for (int r = 0; r < 8; ++r) {
                int row = mt * 16 + hf * 8 + r;
                dst[row * QSTR + colg] = (h16)(acc[r] + bv);
            }
        }
        __syncthreads();

        // ---- stage 2: S = q @ k^T (raw, f32) ; 16 tiles, 2 per wave -----
#pragma unroll
        for (int s = 0; s < 2; ++s) {
            int tt = w * 2 + s;           // 0..15
            int it = tt >> 2, jt = tt & 3;
            v16h A = load_a_h(qh + it * 16 * QSTR, QSTR, lane);    // K = 32 = head dim
            v16h B = load_b_nk_h(kh + jt * 16 * QSTR, QSTR, lane); // k rows are N
            v8f acc = {};
            acc = WMMA(A, B, acc);
            int hf = lane >> 4, col = jt * 16 + (lane & 15);
#pragma unroll
            for (int r = 0; r < 8; ++r)
                S[(it * 16 + hf * 8 + r) * SSTR + col] = acc[r];
        }
        __syncthreads();

        // ---- stage 3: fused scale + bias + softmax (fp32) ---------------
        {
            int row = t >> 2;
            int c0  = (t & 3) * 16;
            const float* brow = biasp + h * (NTOK * NTOK) + row * NTOK + c0;
            float vals[16];
            float mx = -1e30f;
#pragma unroll
            for (int j = 0; j < 16; ++j) {
                float sv = S[row * SSTR + c0 + j] * scale + brow[j];
                vals[j] = sv;
                mx = fmaxf(mx, sv);
            }
            mx = fmaxf(mx, __shfl_xor(mx, 1, 32));
            mx = fmaxf(mx, __shfl_xor(mx, 2, 32));
            float sum = 0.f;
#pragma unroll
            for (int j = 0; j < 16; ++j) {
                vals[j] = __expf(vals[j] - mx);
                sum += vals[j];
            }
            sum += __shfl_xor(sum, 1, 32);
            sum += __shfl_xor(sum, 2, 32);
            float inv = 1.0f / sum;
#pragma unroll
            for (int j = 0; j < 16; ++j)
                S[row * SSTR + c0 + j] = vals[j] * inv;
        }
        __syncthreads();

        // ---- stage 4: oh = P @ v ; 8 tiles, 1 per wave ------------------
        {
            int mt = w >> 1, nt = w & 1;
            v8f acc = {};
#pragma unroll
            for (int ks = 0; ks < 2; ++ks) {
                v16h A = load_a_f(S + mt * 16 * SSTR + ks * 32, SSTR, lane);  // P rows
                v16h B = load_b_kn_h(vh + ks * 32 * QSTR + nt * 16, QSTR, lane);
                acc = WMMA(A, B, acc);
            }
            int hf = lane >> 4, col = nt * 16 + (lane & 15);
#pragma unroll
            for (int r = 0; r < 8; ++r)
                oh[(mt * 16 + hf * 8 + r) * QSTR + col] = (h16)acc[r];
        }
        __syncthreads();

        // ---- stage 5: fold head into projection accumulators ------------
        // out_tile(mt, n) += oh_tile(mt) @ proj_w[n*16.., h*32..]^T ; K=32.
        {
            int mt  = w & 3;
            int ntb = (w >> 2) * 8;
            v16h A = load_a_h(oh + mt * 16 * QSTR, QSTR, lane);
#pragma unroll
            for (int j = 0; j < 8; ++j) {
                v16h B = ppack[((ntb + j) * 8 + h) * 32 + lane]; // packed f16
                pacc[j] = WMMA(A, B, pacc[j]);
            }
        }
        __syncthreads();  // qh/kh/vh/S/oh reused next head
    }

    // ---- epilogue: add proj bias, write fp32 output ----------------------
    {
        int mt  = w & 3;
        int ntb = (w >> 2) * 8;
        int hf  = lane >> 4, c16 = lane & 15;
        float* ob = out + (size_t)b * NTOK * DIM;
#pragma unroll
        for (int j = 0; j < 8; ++j) {
            int col = (ntb + j) * 16 + c16;
            float pb = proj_b[col];
#pragma unroll
            for (int r = 0; r < 8; ++r) {
                int row = mt * 16 + hf * 8 + r;
                ob[row * DIM + col] = pacc[j][r] + pb;
            }
        }
    }
}

extern "C" void kernel_launch(void* const* d_in, const int* in_sizes, int n_in,
                              void* d_out, int out_size, void* d_ws, size_t ws_size,
                              hipStream_t stream) {
    const float* x          = (const float*)d_in[0];
    const float* qkv_w      = (const float*)d_in[1];
    const float* qkv_b      = (const float*)d_in[2];
    const float* proj_w     = (const float*)d_in[3];
    const float* proj_b     = (const float*)d_in[4];
    const float* bias_table = (const float*)d_in[5];
    const int*   rel_idx    = (const int*)d_in[6];
    float* out = (float*)d_out;

    // Pre-pass: pack weights to f16 fragments + materialize bias (640 KB).
    int pack_items = (QKV_FRAGS + PROJ_FRAGS) * 32 + NHEADS * NTOK * NTOK;
    int pack_blocks = (pack_items + 255) / 256;
    pack_kernel<<<pack_blocks, 256, 0, stream>>>(qkv_w, proj_w, bias_table,
                                                 rel_idx, d_ws);

    int nwin = in_sizes[0] / (NTOK * DIM);  // 8192
    lewin_attn_kernel<<<nwin, 256, SMEM_TOT, stream>>>(
        x, qkv_b, proj_b, d_ws, out);
}
